// TransformerBlock_28424093565577
// MI455X (gfx1250) — matrix operand order repaired and checked by reading, not compiled, not run on hardware
//
#include <hip/hip_runtime.h>
#include <cmath>

typedef unsigned short u16;
typedef __attribute__((ext_vector_type(8)))  float  v8f;
typedef __attribute__((ext_vector_type(16))) __bf16 bf16x16;

union Frag16 { uint4 u[2]; bf16x16 v; };
static_assert(sizeof(Frag16) == 32, "frag size");

__device__ __forceinline__ u16 f2bf(float f) {
    unsigned u = __builtin_bit_cast(unsigned, f);
    unsigned r = u + 0x7FFFu + ((u >> 16) & 1u);   // round-to-nearest-even
    return (u16)(r >> 16);
}

// CDNA5 async global->LDS copy, 16 bytes per lane, tracked by ASYNCcnt.
// LDS operand: low 32 bits of the flat aperture address == LDS byte address.
__device__ __forceinline__ void async_b128(const u16* lds_generic, const u16* gptr) {
    unsigned ldsa = (unsigned)(uintptr_t)lds_generic;
    unsigned long long ga = (unsigned long long)(uintptr_t)gptr;
    asm volatile("global_load_async_to_lds_b128 %0, %1, off"
                 :: "v"(ldsa), "v"(ga) : "memory");
}
__device__ __forceinline__ void wait_async_le4() {
    asm volatile("s_wait_asynccnt 0x4" ::: "memory");
}
__device__ __forceinline__ void wait_async_0() {
    asm volatile("s_wait_asynccnt 0x0" ::: "memory");
}

// ---------------------------------------------------------------------------
// Weight prep: W[K,N] f32 row-major  ->  Wt[N,K] bf16 row-major (one pass)
// ---------------------------------------------------------------------------
__global__ __launch_bounds__(256) void wtrans_kernel(
    const float* __restrict__ W, u16* __restrict__ Wt, int K, int N)
{
    __shared__ u16 t[64 * 66];
    const int k0 = blockIdx.y * 64, n0 = blockIdx.x * 64;
    const int tid = threadIdx.x;
    #pragma unroll
    for (int i = 0; i < 16; ++i) {
        const int idx = tid + (i << 8);
        const int k = idx >> 6, n = idx & 63;
        t[n * 66 + k] = f2bf(W[(size_t)(k0 + k) * N + n0 + n]);
    }
    __syncthreads();
    #pragma unroll
    for (int i = 0; i < 16; ++i) {
        const int idx = tid + (i << 8);
        const int n = idx >> 6, k = idx & 63;
        Wt[(size_t)(n0 + n) * K + k0 + k] = t[n * 66 + k];
    }
}

// ---------------------------------------------------------------------------
// WMMA GEMM:  C[M,N] = A_bf16[M,K] @ Bt_bf16[N,K]^T  (+ epilogue)
// 128x128 block tile, 8 waves x (32x64) wave tiles, BK=32.
// Double-buffered LDS, tiles streamed by async global->LDS copies (peeled
// pipeline: no branches in the steady state).
//  EPI: 0 = +bias -> bf16             (Q/K projections)
//       1 = *scale, mask -> f32       (attention scores)
//       2 = plain -> bf16             (attn @ V)
//       3 = +bias, relu -> bf16       (FFN1)
//       4 = +bias +resid -> f32       (O-proj / FFN2, fused residual)
//       5 = +bias -> bf16 transposed  (V projection -> V^T [E,S] per batch)
// ---------------------------------------------------------------------------
template<int EPI>
__global__ __launch_bounds__(256) void gemm_wmma(
    const u16* __restrict__ A, const u16* __restrict__ B,
    const float* __restrict__ bias, const int* __restrict__ mask,
    const float* __restrict__ resid,
    float* __restrict__ outF, u16* __restrict__ outB,
    int M, int N, int K, float scale,
    long long sA, long long sB, long long sO, long long sM)
{
    constexpr int BM = 128, BN = 128, BK = 32;
    constexpr int AS = 40, BS = 40;  // padded LDS strides (halfwords) = 80B rows
    __shared__ __align__(16) u16 Asmem[2][BM * AS];
    __shared__ __align__(16) u16 Bsmem[2][BN * BS];

    const int tid  = threadIdx.x;
    const int lane = tid & 31, wave = tid >> 5;
    const int wm = wave >> 1, wn = wave & 1;   // 4x2 waves -> 128(M) x 128(N)
    const int l16 = lane & 15, hsel = lane >> 4;
    const int m0 = blockIdx.y * BM, n0 = blockIdx.x * BN;
    const int z  = blockIdx.z;

    A += (size_t)z * sA;
    B += (size_t)z * sB;
    if (mask) mask += (size_t)z * sM;

    // staging coordinates (per thread): A = 2x b128, B = 2x b128
    const int arow = tid >> 1, acol = (tid & 1) << 4;
    const u16* gA = A + (size_t)(m0 + arow) * K + acol;
    const u16* gB = B + (size_t)(n0 + arow) * K + acol;

    v8f c[2][4];
    #pragma unroll
    for (int i = 0; i < 2; ++i)
        #pragma unroll
        for (int j = 0; j < 4; ++j) c[i][j] = (v8f)0.0f;

    const int nk = K / BK;

    auto stage = [&](int kt, int buf) {
        const int ko = kt * BK;
        async_b128(&Asmem[buf][arow * AS + acol],     gA + ko);
        async_b128(&Asmem[buf][arow * AS + acol + 8], gA + ko + 8);
        async_b128(&Bsmem[buf][arow * BS + acol],     gB + ko);
        async_b128(&Bsmem[buf][arow * BS + acol + 8], gB + ko + 8);
    };
    auto compute = [&](int buf) {
        Frag16 a[2], b[4];
        #pragma unroll
        for (int i = 0; i < 2; ++i) {
            const int row = wm * 32 + i * 16 + l16;             // M = lane%16
            const uint4* p = (const uint4*)&Asmem[buf][row * AS + (hsel << 3)];
            a[i].u[0] = p[0];                                   // K = 8h..8h+7
            a[i].u[1] = p[2];                                   // K = 16+8h..
        }
        #pragma unroll
        for (int j = 0; j < 4; ++j) {
            const int col = wn * 64 + j * 16 + l16;             // N = lane%16
            const uint4* p = (const uint4*)&Bsmem[buf][col * BS + (hsel << 3)];
            b[j].u[0] = p[0];                                   // K = 8h..8h+7
            b[j].u[1] = p[2];
        }
        #pragma unroll
        for (int i = 0; i < 2; ++i)
            #pragma unroll
            for (int j = 0; j < 4; ++j)
                c[i][j] = __builtin_amdgcn_wmma_f32_16x16x32_bf16(
                    false, a[i].v, false, b[j].v, (short)0, c[i][j], false, false);
    };

    // ---- peeled double-buffered async pipeline (branch-free steady state) --
    stage(0, 0);
    for (int kk = 0; kk < nk - 1; ++kk) {
        stage(kk + 1, (kk + 1) & 1);
        wait_async_le4();               // retire tile kk (in-order completion)
        __syncthreads();                // tile kk visible to all waves
        compute(kk & 1);
        __syncthreads();                // reads done before buffer overwrite
    }
    wait_async_0();
    __syncthreads();
    compute((nk - 1) & 1);

    // ---- epilogue: C/D layout -> lanes 0-15: M=vgpr, lanes 16-31: M=8+vgpr --
    float* oF = outF ? outF + (size_t)z * sO : nullptr;
    u16*   oB = outB ? outB + (size_t)z * sO : nullptr;
    #pragma unroll
    for (int i = 0; i < 2; ++i) {
        const int mBase = m0 + wm * 32 + i * 16 + (hsel << 3);
        #pragma unroll
        for (int j = 0; j < 4; ++j) {
            const int nCol = n0 + wn * 64 + j * 16 + l16;
            float bv = 0.0f;
            if (EPI == 0 || EPI == 3 || EPI == 4 || EPI == 5) bv = bias[nCol];
            if (EPI == 5) {
                // transposed store: out[nCol][mBase..mBase+7], packed b128
                unsigned wd[4];
                #pragma unroll
                for (int p = 0; p < 4; ++p)
                    wd[p] = (unsigned)f2bf(c[i][j][2 * p] + bv) |
                            ((unsigned)f2bf(c[i][j][2 * p + 1] + bv) << 16);
                uint4 pk; pk.x = wd[0]; pk.y = wd[1]; pk.z = wd[2]; pk.w = wd[3];
                *(uint4*)&oB[(size_t)nCol * M + mBase] = pk;
            } else {
                #pragma unroll
                for (int v = 0; v < 8; ++v) {
                    const int m = mBase + v;
                    const size_t off = (size_t)m * N + nCol;
                    const float acc = c[i][j][v];
                    if (EPI == 0) {
                        oB[off] = f2bf(acc + bv);
                    } else if (EPI == 1) {
                        float sv = acc * scale;
                        if (mask[nCol] == 0) sv = -1e9f;
                        oF[off] = sv;
                    } else if (EPI == 2) {
                        oB[off] = f2bf(acc);
                    } else if (EPI == 3) {
                        float r = acc + bv;
                        oB[off] = f2bf(r > 0.0f ? r : 0.0f);
                    } else if (EPI == 4) {
                        oF[off] = acc + bv + resid[off];
                    }
                }
            }
        }
    }
}

// ---------------------------------------------------------------------------
// Block reductions
// ---------------------------------------------------------------------------
__device__ __forceinline__ float blkSum(float v, float* red, int tid) {
    red[tid] = v; __syncthreads();
    for (int s = 128; s > 0; s >>= 1) {
        if (tid < s) red[tid] += red[tid + s];
        __syncthreads();
    }
    float r = red[0]; __syncthreads(); return r;
}
__device__ __forceinline__ float blkMax(float v, float* red, int tid) {
    red[tid] = v; __syncthreads();
    for (int s = 128; s > 0; s >>= 1) {
        if (tid < s) red[tid] = fmaxf(red[tid], red[tid + s]);
        __syncthreads();
    }
    float r = red[0]; __syncthreads(); return r;
}

// LayerNorm: one block per row, output bf16
__global__ __launch_bounds__(256) void ln_kernel(
    const float* __restrict__ x, const float* __restrict__ g,
    const float* __restrict__ b, u16* __restrict__ out, int E)
{
    __shared__ float red[256];
    const int tid = threadIdx.x;
    const float* xr = x + (size_t)blockIdx.x * E;
    float s = 0.0f, sq = 0.0f;
    for (int c = tid; c < E; c += 256) { float v = xr[c]; s += v; sq += v * v; }
    s  = blkSum(s,  red, tid);
    sq = blkSum(sq, red, tid);
    const float mean = s / E;
    const float var  = sq / E - mean * mean;
    const float rstd = rsqrtf(var + 1e-5f);
    u16* o = out + (size_t)blockIdx.x * E;
    for (int c = tid; c < E; c += 256)
        o[c] = f2bf((xr[c] - mean) * rstd * g[c] + b[c]);
}

// Softmax: in-place on f32 scores row, also emits bf16 copy for attn@V
__global__ __launch_bounds__(256) void softmax_kernel(
    float* __restrict__ sc, u16* __restrict__ attnB, int S)
{
    __shared__ float red[256];
    const int tid = threadIdx.x;
    float* r = sc + (size_t)blockIdx.x * S;
    float m = -3.4e38f;
    for (int c = tid; c < S; c += 256) m = fmaxf(m, r[c]);
    m = blkMax(m, red, tid);
    float s = 0.0f;
    for (int c = tid; c < S; c += 256) s += __expf(r[c] - m);
    s = blkSum(s, red, tid);
    const float inv = 1.0f / s;
    u16* ab = attnB + (size_t)blockIdx.x * S;
    for (int c = tid; c < S; c += 256) {
        float v = __expf(r[c] - m) * inv;
        r[c] = v; ab[c] = f2bf(v);
    }
}

// ---------------------------------------------------------------------------
extern "C" void kernel_launch(void* const* d_in, const int* in_sizes, int n_in,
                              void* d_out, int out_size, void* d_ws, size_t ws_size,
                              hipStream_t stream)
{
    const float* x   = (const float*)d_in[0];
    const int*   msk = (const int*)  d_in[1];
    const float* Wq  = (const float*)d_in[2];  const float* bq = (const float*)d_in[3];
    const float* Wk  = (const float*)d_in[4];  const float* bk = (const float*)d_in[5];
    const float* Wv  = (const float*)d_in[6];  const float* bv = (const float*)d_in[7];
    const float* Wo  = (const float*)d_in[8];  const float* bo = (const float*)d_in[9];
    const float* W1  = (const float*)d_in[10]; const float* b1 = (const float*)d_in[11];
    const float* W2  = (const float*)d_in[12]; const float* b2 = (const float*)d_in[13];
    const float* g1  = (const float*)d_in[14]; const float* be1= (const float*)d_in[15];
    const float* g2  = (const float*)d_in[16]; const float* be2= (const float*)d_in[17];

    const long long E   = in_sizes[3];
    const long long DFF = in_sizes[11];
    const long long BS  = in_sizes[1];                    // B*S rows
    const long long S   = ((long long)out_size - BS * E) / BS;
    const long long Bn  = BS / S;

    // workspace carve-out (256B aligned)
    char* w = (char*)d_ws;
    auto carve = [&](size_t bytes) { char* p = w; w += (bytes + 255) & ~(size_t)255; return p; };
    u16*   xn    = (u16*)  carve((size_t)BS * E * 2);     // reused for xn2
    u16*   q     = (u16*)  carve((size_t)BS * E * 2);     // reused for ctx
    u16*   kmat  = (u16*)  carve((size_t)BS * E * 2);
    u16*   vT    = (u16*)  carve((size_t)BS * E * 2);     // [B][E][S]
    u16*   attnB = (u16*)  carve((size_t)Bn * S * S * 2);
    float* x2    = (float*)carve((size_t)BS * E * 4);
    u16*   h     = (u16*)  carve((size_t)BS * DFF * 2);
    u16*   Wqt   = (u16*)  carve((size_t)E * E * 2);      // bf16 [N,K] weights
    u16*   Wkt   = (u16*)  carve((size_t)E * E * 2);
    u16*   Wvt   = (u16*)  carve((size_t)E * E * 2);
    u16*   Wot   = (u16*)  carve((size_t)E * E * 2);
    u16*   W1t   = (u16*)  carve((size_t)E * DFF * 2);
    u16*   W2t   = (u16*)  carve((size_t)E * DFF * 2);
    u16* ctx = q;
    u16* xn2 = xn;

    float* outMain = (float*)d_out;
    float* scoresF = outMain + (size_t)BS * E;            // attn output region

    const dim3 blk(256);
    const float scl = 1.0f / sqrtf((float)E);

    // 0) one-shot weight convert + transpose to bf16 [N,K]
    dim3 gtE((unsigned)(E / 64), (unsigned)(E / 64), 1);
    wtrans_kernel<<<gtE, blk, 0, stream>>>(Wq, Wqt, (int)E, (int)E);
    wtrans_kernel<<<gtE, blk, 0, stream>>>(Wk, Wkt, (int)E, (int)E);
    wtrans_kernel<<<gtE, blk, 0, stream>>>(Wv, Wvt, (int)E, (int)E);
    wtrans_kernel<<<gtE, blk, 0, stream>>>(Wo, Wot, (int)E, (int)E);
    dim3 gt1((unsigned)(DFF / 64), (unsigned)(E / 64), 1);   // W1 [E,DFF] -> [DFF,E]
    wtrans_kernel<<<gt1, blk, 0, stream>>>(W1, W1t, (int)E, (int)DFF);
    dim3 gt2((unsigned)(E / 64), (unsigned)(DFF / 64), 1);   // W2 [DFF,E] -> [E,DFF]
    wtrans_kernel<<<gt2, blk, 0, stream>>>(W2, W2t, (int)DFF, (int)E);

    // 1) pre-LN
    ln_kernel<<<dim3((unsigned)BS), blk, 0, stream>>>(x, g1, be1, xn, (int)E);

    // 2) Q/K projections (bf16 out, fused bias); V stored transposed per batch
    dim3 gqkv((unsigned)(E / 128), (unsigned)(BS / 128), 1);
    gemm_wmma<0><<<gqkv, blk, 0, stream>>>(xn, Wqt, bq, nullptr, nullptr,
        nullptr, q,    (int)BS, (int)E, (int)E, 1.0f, 0, 0, 0, 0);
    gemm_wmma<0><<<gqkv, blk, 0, stream>>>(xn, Wkt, bk, nullptr, nullptr,
        nullptr, kmat, (int)BS, (int)E, (int)E, 1.0f, 0, 0, 0, 0);
    dim3 gv((unsigned)(E / 128), (unsigned)(S / 128), (unsigned)Bn);
    gemm_wmma<5><<<gv, blk, 0, stream>>>(xn, Wvt, bv, nullptr, nullptr,
        nullptr, vT, (int)S, (int)E, (int)E, 1.0f, S * E, 0, E * S, 0);

    // 3) scores = q @ k^T * 1/sqrt(E), masked -> f32 directly into d_out attn
    dim3 gsc((unsigned)(S / 128), (unsigned)(S / 128), (unsigned)Bn);
    gemm_wmma<1><<<gsc, blk, 0, stream>>>(q, kmat, nullptr, msk, nullptr,
        scoresF, nullptr, (int)S, (int)S, (int)E, scl, S * E, S * E, S * S, S);

    // 4) softmax rows (in-place f32 + bf16 copy)
    softmax_kernel<<<dim3((unsigned)BS), blk, 0, stream>>>(scoresF, attnB, (int)S);

    // 5) ctx = attn @ v   (B = V^T [E,S] bf16, direct staging)
    dim3 gav((unsigned)(E / 128), (unsigned)(S / 128), (unsigned)Bn);
    gemm_wmma<2><<<gav, blk, 0, stream>>>(attnB, vT, nullptr, nullptr,
        nullptr, nullptr, ctx, (int)S, (int)E, (int)S, 1.0f, S * S, E * S, S * E, 0);

    // 6) x2 = ctx @ Wo + bo + x  (fused residual, f32)
    dim3 go((unsigned)(E / 128), (unsigned)(BS / 128), 1);
    gemm_wmma<4><<<go, blk, 0, stream>>>(ctx, Wot, bo, nullptr, x,
        x2, nullptr, (int)BS, (int)E, (int)E, 1.0f, 0, 0, 0, 0);

    // 7) LN2
    ln_kernel<<<dim3((unsigned)BS), blk, 0, stream>>>(x2, g2, be2, xn2, (int)E);

    // 8) h = relu(xn2 @ W1 + b1)  (bf16)
    dim3 gf1((unsigned)(DFF / 128), (unsigned)(BS / 128), 1);
    gemm_wmma<3><<<gf1, blk, 0, stream>>>(xn2, W1t, b1, nullptr, nullptr,
        nullptr, h, (int)BS, (int)DFF, (int)E, 1.0f, 0, 0, 0, 0);

    // 9) out = h @ W2 + b2 + x2  (fused residual, f32 -> d_out main)
    dim3 gf2((unsigned)(E / 128), (unsigned)(BS / 128), 1);
    gemm_wmma<4><<<gf2, blk, 0, stream>>>(h, W2t, b2, nullptr, x2,
        outMain, nullptr, (int)BS, (int)E, (int)DFF, 1.0f, 0, 0, 0, 0);
}
